// CMA_45732811768249
// MI455X (gfx1250) — compile-verified, matrix operand order checked by reading, not run on hardware
//
#include <hip/hip_runtime.h>
#include <hip/hip_bf16.h>

// ---------------------------------------------------------------------------
// Problem constants (from reference)
// ---------------------------------------------------------------------------
#define BB 1024
#define DD 512
#define TT 36
#define NTOK 64
#define DSC 32
#define NG 4
#define EPSF 1e-5f

typedef __attribute__((ext_vector_type(16))) __bf16 v16bf;
typedef __attribute__((ext_vector_type(8)))  float  v8f;

// ---------------------------------------------------------------------------
// WMMA helpers — 16-bit A/B fragment lane/VGPR striping per CDNA5 ISA 7.12.2:
//   lanes 0-15 : row = lane,    VGPR0-3 hold K = 0..7 (packed pairs), VGPR4-7 K=16..23
//   lanes 16-31: row = lane-16, VGPR0-3 hold K = 8..15,               VGPR4-7 K=24..31
// koff(e) for element e of the v16bf (pairs contiguous in K):
//   ((e>=8)?16:0) + hi*8 + ((e>>1)&3)*2 + (e&1)
// ---------------------------------------------------------------------------
__device__ __forceinline__ v8f wmma_bf16(v16bf a, v16bf b, v8f c) {
  return __builtin_amdgcn_wmma_f32_16x16x32_bf16(false, a, false, b,
                                                 (short)0, c, false, false);
}

// fragment from a row-major [R][K] bf16 LDS buffer: p[(r0+l)*ld + kb + koff]
__device__ __forceinline__ v16bf frag_rm_bf(const __bf16* p, int ld, int r0,
                                            int kb, int l, int hi) {
  v16bf f;
  const __bf16* row = p + (r0 + l) * ld + kb + hi * 8;
#pragma unroll
  for (int e = 0; e < 16; ++e)
    f[e] = row[((e >> 3) << 4) + ((e >> 1) & 3) * 2 + (e & 1)];
  return f;
}

// fragment from a K-major [K][R] bf16 LDS buffer: p[(kb+koff)*ld + r0 + l]
__device__ __forceinline__ v16bf frag_km_bf(const __bf16* p, int ld, int r0,
                                            int kb, int l, int hi) {
  v16bf f;
  const __bf16* col = p + (kb + hi * 8) * ld + r0 + l;
#pragma unroll
  for (int e = 0; e < 16; ++e)
    f[e] = col[(((e >> 3) << 4) + ((e >> 1) & 3) * 2 + (e & 1)) * ld];
  return f;
}

// fragment from a row-major [R][K] f32 GLOBAL buffer (convert to bf16)
__device__ __forceinline__ v16bf frag_rm_f32g(const float* p, int ld, int r0,
                                              int kb, int l, int hi) {
  v16bf f;
  const float* row = p + (r0 + l) * ld + kb + hi * 8;
#pragma unroll
  for (int e = 0; e < 16; ++e)
    f[e] = (__bf16)row[((e >> 3) << 4) + ((e >> 1) & 3) * 2 + (e & 1)];
  return f;
}

// fragment from a K-major [K][R] f32 GLOBAL buffer, row clamped (padding rows
// re-read row rmax; their outputs are discarded downstream)
__device__ __forceinline__ v16bf frag_km_f32g_clamp(const float* p, int ld,
                                                    int r0, int kb, int rmax,
                                                    int l, int hi) {
  v16bf f;
  int r = r0 + l;
  r = (r > rmax) ? rmax : r;
  const float* col = p + (kb + hi * 8) * ld + r;
#pragma unroll
  for (int e = 0; e < 16; ++e)
    f[e] = (__bf16)col[(((e >> 3) << 4) + ((e >> 1) & 3) * 2 + (e & 1)) * ld];
  return f;
}

__device__ __forceinline__ float wave_sum(float v) {
#pragma unroll
  for (int o = 16; o; o >>= 1) v += __shfl_xor(v, o, 32);
  return v;
}

// ---------------------------------------------------------------------------
// LDS layout for the attention kernel (dynamic LDS, 187392 B of 320 KB/WGP)
//   ysb  bf16 [512][64]   @ 0       (dead after M2)
//   S1   f32  [64][48]    @ 65536   (dead after softmax1)
//   xc   f32  [36][520]   @ 0       (aliases ysb+S1; written after both dead)
//   att1 bf16 [64][64]    @ 77824
//   S2   f32  [48][64]    @ 86016
//   att2 bf16 [48][64]    @ 98304
//   rep  bf16 [64][520]   @ 104448
//   hrow f32  [8][512]    @ 171008
// ---------------------------------------------------------------------------
#define YS_OFF   0
#define S1_OFF   65536
#define ATT1_OFF 77824
#define S2_OFF   86016
#define ATT2_OFF 98304
#define REP_OFF  104448
#define HROW_OFF 171008
#define SMEM1    187392

__global__ void __launch_bounds__(256, 1)
cma_attn_kernel(const float* __restrict__ x, const float* __restrict__ y,
                const float* __restrict__ tokens,
                const float* __restrict__ gate_av,
                const float* __restrict__ w_down,
                const float* __restrict__ lnb_g, const float* __restrict__ lnb_b,
                float* __restrict__ z_buf) {
  extern __shared__ char smem[];
  __bf16* ysb  = (__bf16*)(smem + YS_OFF);
  float*  S1   = (float*)(smem + S1_OFF);
  float*  xc   = (float*)(smem + YS_OFF);   // alias (phase-separated)
  __bf16* att1 = (__bf16*)(smem + ATT1_OFF);
  float*  S2   = (float*)(smem + S2_OFF);
  __bf16* att2 = (__bf16*)(smem + ATT2_OFF);
  __bf16* rep  = (__bf16*)(smem + REP_OFF);
  float*  hrow = (float*)(smem + HROW_OFF);

  const int b    = blockIdx.x;
  const int tid  = threadIdx.x;
  const int w    = tid >> 5;
  const int lane = tid & 31;
  const int l    = lane & 15;
  const int hi   = lane >> 4;

  const float* xs   = x + (size_t)b * (DD * TT);
  const float* ysrc = y + (size_t)b * (DD * TT);
  const float  gav  = gate_av[0];

  // ---- P0: stage y[b] into LDS as bf16 [d][t] (t padded 36->64 with zeros),
  //          prefetch x[b] (global_prefetch_b8) for the M3 / residual reads.
  for (int i = tid; i < (DD * TT * 4) / 256; i += 256)
    __builtin_prefetch((const char*)xs + i * 256, 0, 1);
  for (int i = tid; i < DD * 64; i += 256) {
    int d = i >> 6, t = i & 63;
    float v = (t < TT) ? ysrc[d * TT + t] : 0.f;
    ysb[i] = (__bf16)v;
  }
  __syncthreads();

  // ---- P1: S1[64 tok][48 t] = tokens[64,512] x y[512,36]  (K = 512)
  for (int tt = w; tt < 12; tt += 8) {
    int mt = tt / 3, nt = tt % 3;
    v8f acc = {};
    for (int kb = 0; kb < DD; kb += 32) {
      v16bf a  = frag_rm_f32g(tokens, DD, mt * 16, kb, l, hi);
      v16bf bf = frag_km_bf(ysb, 64, nt * 16, kb, l, hi);
      acc = wmma_bf16(a, bf, acc);
    }
#pragma unroll
    for (int e = 0; e < 8; ++e)
      S1[(mt * 16 + e + hi * 8) * 48 + nt * 16 + l] = acc[e];
  }
  __syncthreads();

  // ---- softmax over t (36 valid) per token row; pad cols -> 0
  if (tid < NTOK) {
    const float* r = S1 + tid * 48;
    float m = r[0];
    for (int t = 1; t < TT; ++t) m = fmaxf(m, r[t]);
    float s = 0.f;
    float e36[TT];
    for (int t = 0; t < TT; ++t) { e36[t] = __expf(r[t] - m); s += e36[t]; }
    float inv = 1.f / s;
    for (int t = 0; t < TT; ++t) att1[tid * 64 + t] = (__bf16)(e36[t] * inv);
    for (int t = TT; t < 64; ++t) att1[tid * 64 + t] = (__bf16)0.f;
  }
  __syncthreads();

  // ---- P2: rep[64 tok][512 d] = tokens + att1[64,64] x yT[64,512] (K = 64)
  for (int tt = 0; tt < 16; ++tt) {
    int ti = w * 16 + tt;
    int mt = ti >> 5, nt = ti & 31;
    v8f acc = {};
    for (int kb = 0; kb < 64; kb += 32) {
      v16bf a  = frag_rm_bf(att1, 64, mt * 16, kb, l, hi);
      v16bf bf = frag_rm_bf(ysb, 64, nt * 16, kb, l, hi);  // ys[d][t]=[n][k]
      acc = wmma_bf16(a, bf, acc);
    }
#pragma unroll
    for (int e = 0; e < 8; ++e) {
      int m = mt * 16 + e + hi * 8;
      int n = nt * 16 + l;
      rep[m * 520 + n] = (__bf16)(acc[e] + tokens[m * DD + n]);
    }
  }
  __syncthreads();

  // ---- P3: S2[48 t][64 tok] = xT[36,512] x repT[512,64]  (K = 512)
  for (int tt = w; tt < 12; tt += 8) {
    int mt = tt >> 2, nt = tt & 3;
    v8f acc = {};
    for (int kb = 0; kb < DD; kb += 32) {
      v16bf a  = frag_km_f32g_clamp(xs, TT, mt * 16, kb, TT - 1, l, hi);
      v16bf bf = frag_rm_bf(rep, 520, nt * 16, kb, l, hi);  // rep[n][k]
      acc = wmma_bf16(a, bf, acc);
    }
#pragma unroll
    for (int e = 0; e < 8; ++e)
      S2[(mt * 16 + e + hi * 8) * 64 + nt * 16 + l] = acc[e];
  }
  __syncthreads();

  // ---- softmax over 64 tokens per valid t row; rows >=36 zeroed
  if (tid < 48) {
    if (tid < TT) {
      const float* r = S2 + tid * 64;
      float m = r[0];
      for (int c = 1; c < NTOK; ++c) m = fmaxf(m, r[c]);
      float s = 0.f;
      float ev[NTOK];
      for (int c = 0; c < NTOK; ++c) { ev[c] = __expf(r[c] - m); s += ev[c]; }
      float inv = 1.f / s;
      for (int c = 0; c < NTOK; ++c) att2[tid * 64 + c] = (__bf16)(ev[c] * inv);
    } else {
      for (int c = 0; c < NTOK; ++c) att2[tid * 64 + c] = (__bf16)0.f;
    }
  }
  __syncthreads();

  // ---- P4: x_res[48 t][512 d] = att2[48,64] x rep[64,512] (K = 64);
  //          fuse residual: xc[t][d] = x[d][t] + gate_av * x_res  (t < 36)
  //          xc aliases ysb/S1 which are dead past the barrier above.
  for (int tt = 0; tt < 12; ++tt) {
    int ti = w * 12 + tt;
    int mt = ti >> 5, nt = ti & 31;
    v8f acc = {};
    for (int kb = 0; kb < 64; kb += 32) {
      v16bf a  = frag_rm_bf(att2, 64, mt * 16, kb, l, hi);
      v16bf bf = frag_km_bf(rep, 520, nt * 16, kb, l, hi);  // rep[k][n]
      acc = wmma_bf16(a, bf, acc);
    }
#pragma unroll
    for (int e = 0; e < 8; ++e) {
      int t = mt * 16 + e + hi * 8;
      if (t < TT) {
        int d = nt * 16 + l;
        xc[t * 520 + d] = xs[d * TT + t] + gav * acc[e];
      }
    }
  }
  __syncthreads();

  // ---- P5: per-t LayerNorm over 512 channels + grouped down conv -> z
  for (int t = w; t < TT; t += 8) {
    float v[16];
    float s = 0.f, ss = 0.f;
#pragma unroll
    for (int j = 0; j < 16; ++j) {
      float vv = xc[t * 520 + lane + 32 * j];
      v[j] = vv; s += vv; ss += vv * vv;
    }
    s = wave_sum(s); ss = wave_sum(ss);
    float mu  = s * (1.f / 512.f);
    float var = ss * (1.f / 512.f) - mu * mu;
    float rs  = rsqrtf(var + EPSF);
#pragma unroll
    for (int j = 0; j < 16; ++j) {
      int d = lane + 32 * j;
      hrow[w * 512 + d] = (v[j] - mu) * rs * lnb_g[d] + lnb_b[d];
    }
    // lane == output bottleneck channel (32 lanes, 32 channels)
    int o = lane;
    int gbase = (o >> 3) * 128;
    float accz = 0.f;
    for (int c = 0; c < 128; ++c)
      accz = fmaf(w_down[o * 128 + c], hrow[w * 512 + gbase + c], accz);
    z_buf[(size_t)b * DSC * TT + o * TT + t] = accz;
  }
}

// ---------------------------------------------------------------------------
// bn1 batch statistics: one block per bottleneck channel (32 blocks)
// ---------------------------------------------------------------------------
__global__ void __launch_bounds__(256)
bn1_stats_kernel(const float* __restrict__ z_buf, float* __restrict__ mu,
                 float* __restrict__ rstd) {
  __shared__ float sred[256], sred2[256];
  int c = blockIdx.x, tid = threadIdx.x;
  float s = 0.f, ss = 0.f;
  for (int i = tid; i < BB * TT; i += 256) {
    int b = i / TT, t = i - b * TT;
    float v = z_buf[(size_t)b * DSC * TT + c * TT + t];
    s += v; ss += v * v;
  }
  sred[tid] = s; sred2[tid] = ss;
  __syncthreads();
  for (int o = 128; o; o >>= 1) {
    if (tid < o) { sred[tid] += sred[tid + o]; sred2[tid] += sred2[tid + o]; }
    __syncthreads();
  }
  if (tid == 0) {
    float ninv = 1.f / (float)(BB * TT);
    float m = sred[0] * ninv;
    mu[c] = m;
    rstd[c] = rsqrtf(sred2[0] * ninv - m * m + EPSF);
  }
}

// ---------------------------------------------------------------------------
// up-conv recompute + bn2 sum accumulation (atomics); one block per batch
// ---------------------------------------------------------------------------
__global__ void __launch_bounds__(256)
up_stats_kernel(const float* __restrict__ z_buf, const float* __restrict__ mu1,
                const float* __restrict__ rstd1, const float* __restrict__ bn1g,
                const float* __restrict__ bn1b, const float* __restrict__ w_up,
                float* __restrict__ sum2, float* __restrict__ ssq2) {
  __shared__ float zr[DSC * TT];
  int b = blockIdx.x, tid = threadIdx.x;
  for (int i = tid; i < DSC * TT; i += 256) {
    int c = i / TT;
    float v = z_buf[(size_t)b * DSC * TT + i];
    v = (v - mu1[c]) * rstd1[c] * bn1g[c] + bn1b[c];
    zr[i] = fmaxf(v, 0.f);
  }
  __syncthreads();
  for (int d = tid; d < DD; d += 256) {
    int gb = (d >> 7) * 8;
    float wu[8];
#pragma unroll
    for (int c = 0; c < 8; ++c) wu[c] = w_up[d * 8 + c];
    float s = 0.f, ss = 0.f;
    for (int t = 0; t < TT; ++t) {
      float o = 0.f;
#pragma unroll
      for (int c = 0; c < 8; ++c) o = fmaf(wu[c], zr[(gb + c) * TT + t], o);
      s += o; ss += o * o;
    }
    atomicAdd(&sum2[d], s);
    atomicAdd(&ssq2[d], ss);
  }
}

// ---------------------------------------------------------------------------
// final: recompute up-conv, bn2 + LayerNorm + gate; coalesced stores via LDS
// ---------------------------------------------------------------------------
#define SMEM3 (DSC * TT * 4 + DD * 37 * 4)

__global__ void __launch_bounds__(256)
out_kernel(const float* __restrict__ z_buf, const float* __restrict__ mu1,
           const float* __restrict__ rstd1, const float* __restrict__ bn1g,
           const float* __restrict__ bn1b, const float* __restrict__ w_up,
           const float* __restrict__ sum2, const float* __restrict__ ssq2,
           const float* __restrict__ bn2g, const float* __restrict__ bn2b,
           const float* __restrict__ lnpg, const float* __restrict__ lnpb,
           const float* __restrict__ gate, float* __restrict__ out) {
  extern __shared__ char smem3[];
  float* zr    = (float*)smem3;                  // [32][36]
  float* stage = (float*)(smem3 + DSC * TT * 4); // [512][37] (pad: no conflicts)
  int b = blockIdx.x, tid = threadIdx.x, w = tid >> 5, lane = tid & 31;
  float g = gate[0];
  for (int i = tid; i < DSC * TT; i += 256) {
    int c = i / TT;
    float v = z_buf[(size_t)b * DSC * TT + i];
    v = (v - mu1[c]) * rstd1[c] * bn1g[c] + bn1b[c];
    zr[i] = fmaxf(v, 0.f);
  }
  __syncthreads();
  const float ninv = 1.f / (float)(BB * TT);
  for (int t = w; t < TT; t += 8) {
    float v[16];
    float s = 0.f, ss = 0.f;
#pragma unroll
    for (int j = 0; j < 16; ++j) {
      int d = lane + 32 * j;
      int gb = (d >> 7) * 8;
      float o = 0.f;
#pragma unroll
      for (int c = 0; c < 8; ++c)
        o = fmaf(w_up[d * 8 + c], zr[(gb + c) * TT + t], o);
      float m2 = sum2[d] * ninv;
      float r2 = rsqrtf(ssq2[d] * ninv - m2 * m2 + EPSF);
      float vv = (o - m2) * r2 * bn2g[d] + bn2b[d];
      v[j] = vv; s += vv; ss += vv * vv;
    }
    s = wave_sum(s); ss = wave_sum(ss);
    float mu  = s * (1.f / 512.f);
    float var = ss * (1.f / 512.f) - mu * mu;
    float rs  = rsqrtf(var + EPSF);
#pragma unroll
    for (int j = 0; j < 16; ++j) {
      int d = lane + 32 * j;
      stage[d * 37 + t] = ((v[j] - mu) * rs * lnpg[d] + lnpb[d]) * g;
    }
  }
  __syncthreads();
  float* ob = out + (size_t)b * DD * TT;
  for (int i = tid; i < DD * TT; i += 256) {
    int d = i / TT, t = i - d * TT;
    ob[i] = stage[d * 37 + t];
  }
}

__global__ void zero_kernel(float* p, int n) {
  int i = blockIdx.x * blockDim.x + threadIdx.x;
  if (i < n) p[i] = 0.f;
}

// ---------------------------------------------------------------------------
extern "C" void kernel_launch(void* const* d_in, const int* in_sizes, int n_in,
                              void* d_out, int out_size, void* d_ws,
                              size_t ws_size, hipStream_t stream) {
  const float* x       = (const float*)d_in[0];
  const float* y       = (const float*)d_in[1];
  const float* tokens  = (const float*)d_in[2];
  const float* gate    = (const float*)d_in[3];
  const float* gate_av = (const float*)d_in[4];
  const float* w_down  = (const float*)d_in[5];
  const float* w_up    = (const float*)d_in[6];
  const float* bn1g    = (const float*)d_in[7];
  const float* bn1b    = (const float*)d_in[8];
  const float* bn2g    = (const float*)d_in[9];
  const float* bn2b    = (const float*)d_in[10];
  const float* lnbg    = (const float*)d_in[11];
  const float* lnbb    = (const float*)d_in[12];
  const float* lnpg    = (const float*)d_in[13];
  const float* lnpb    = (const float*)d_in[14];

  float* ws    = (float*)d_ws;
  float* z_buf = ws;                              // [1024][32][36]
  float* mu1   = ws + (size_t)BB * DSC * TT;      // [32]
  float* rstd1 = mu1 + DSC;                       // [32]
  float* sum2  = rstd1 + DSC;                     // [512]
  float* ssq2  = sum2 + DD;                       // [512]
  float* outp  = (float*)d_out;

  zero_kernel<<<4, 256, 0, stream>>>(sum2, 2 * DD);
  cma_attn_kernel<<<BB, 256, SMEM1, stream>>>(x, y, tokens, gate_av, w_down,
                                              lnbg, lnbb, z_buf);
  bn1_stats_kernel<<<DSC, 256, 0, stream>>>(z_buf, mu1, rstd1);
  up_stats_kernel<<<BB, 256, 0, stream>>>(z_buf, mu1, rstd1, bn1g, bn1b, w_up,
                                          sum2, ssq2);
  out_kernel<<<BB, 256, SMEM3, stream>>>(z_buf, mu1, rstd1, bn1g, bn1b, w_up,
                                         sum2, ssq2, bn2g, bn2b, lnpg, lnpb,
                                         gate, outp);
}
